// FusedLinearJSDLoss_8796093022766
// MI455X (gfx1250) — matrix operand order repaired and checked by reading, not compiled, not run on hardware
//
#include <hip/hip_runtime.h>
#include <stdint.h>

#define BT_DIM 2048
#define H_DIM  2048
#define V_DIM  32000
#define BETA_C 0.5f
#define IGNORE_INDEX (-100)

typedef __attribute__((ext_vector_type(16))) __bf16 v16bf;
typedef __attribute__((ext_vector_type(8)))  float v8f;

union Frag {
    uint32_t u[8];
    uint4    q[2];
    v16bf    v;
};

// ---------------------------------------------------------------------------
// fp32 -> bf16 (round to nearest even), 4 elements per thread
// ---------------------------------------------------------------------------
__device__ __forceinline__ uint16_t f2bf(float f) {
    union { float f; uint32_t u; } c; c.f = f;
    uint32_t u = c.u;
    uint32_t r = (u + 0x7fffu + ((u >> 16) & 1u)) >> 16;
    return (uint16_t)r;
}

__global__ void cvt_f32_bf16(const float* __restrict__ src,
                             uint16_t* __restrict__ dst, int n4) {
    int i = blockIdx.x * blockDim.x + threadIdx.x;
    if (i >= n4) return;
    float4 f = ((const float4*)src)[i];
    ushort4 o;
    o.x = f2bf(f.x); o.y = f2bf(f.y); o.z = f2bf(f.z); o.w = f2bf(f.w);
    ((ushort4*)dst)[i] = o;
}

// ---------------------------------------------------------------------------
// gfx1250 async global->LDS copy (16B per lane), tracked by ASYNCcnt.
// GVS mode: 64-bit SGPR base + per-lane 32-bit byte offset.
// LDS destination address = low 32 bits of generic __shared__ pointer.
// ---------------------------------------------------------------------------
__device__ __forceinline__ void async_copy_b128(void* lds_ptr,
                                                uint64_t base,
                                                uint32_t byte_off) {
    uint32_t lds_addr = (uint32_t)(uintptr_t)lds_ptr;
    asm volatile("global_load_async_to_lds_b128 %0, %1, %2"
                 :: "v"(lds_addr), "v"(byte_off), "s"(base)
                 : "memory");
}

// ---------------------------------------------------------------------------
// C[M,V] = X[M,H] * W[V,H]^T   (bf16 in, fp32 accumulate via WMMA)
// block tile 128x128, k-step 32; 8 waves, wave tile 64x32 (4x2 WMMA tiles)
// double-buffered LDS fed by global_load_async_to_lds_b128
// gridDim = (V/128, M/128, 2)  z: 0 = student, 1 = teacher
// ---------------------------------------------------------------------------
#define LDT 40  // padded LDS row stride in uint16 (80B, 16B aligned, bank-conflict free)
#define KTILES (H_DIM / 32)

__global__ __launch_bounds__(256) void gemm_bf16(
    const uint16_t* __restrict__ Xs, const uint16_t* __restrict__ Ws, float* __restrict__ Cs,
    const uint16_t* __restrict__ Xt, const uint16_t* __restrict__ Wt, float* __restrict__ Ct)
{
    const uint16_t* A; const uint16_t* B; float* C;
    if (blockIdx.z == 0) { A = Xs; B = Ws; C = Cs; }
    else                 { A = Xt; B = Wt; C = Ct; }

    __shared__ __align__(16) uint16_t As[2][128 * LDT];
    __shared__ __align__(16) uint16_t Bs[2][128 * LDT];

    const int tid  = threadIdx.x;
    const int lane = tid & 31;
    const int wave = tid >> 5;
    const int wm   = wave >> 2;   // 0..1  (M direction, 64 rows each)
    const int wn   = wave & 3;    // 0..3  (N direction, 32 cols each)
    const int half = lane >> 4;   // 0/1 lane half
    const int lr   = lane & 15;

    const int blockN = blockIdx.x * 128;
    const int blockM = blockIdx.y * 128;

    // per-thread tile-load coordinates: chunk c in [0,512): row=c/4, col=(c%4)*8
    const int row0 = tid >> 2;            // chunk 0 row
    const int cc0  = (tid & 3) << 3;      // chunk 0 col (bf16 elements)
    const int row1 = row0 + 64;           // chunk 1 (c = tid + 256)
    const int cc1  = cc0;

    const uint64_t baseA = (uint64_t)(uintptr_t)A;
    const uint64_t baseB = (uint64_t)(uintptr_t)B;

    const v8f vzero = {};
    v8f acc[4][2];
#pragma unroll
    for (int mt = 0; mt < 4; ++mt)
#pragma unroll
        for (int nt = 0; nt < 2; ++nt)
            acc[mt][nt] = vzero;

    // ---- issue a k-tile's async copies into LDS buffer `buf`
    auto issue_tile = [&](int k0, int buf) {
        uint32_t a0 = (uint32_t)(((blockM + row0) * H_DIM + k0 + cc0) * 2);
        uint32_t b0 = (uint32_t)(((blockN + row0) * H_DIM + k0 + cc0) * 2);
        uint32_t a1 = (uint32_t)(((blockM + row1) * H_DIM + k0 + cc1) * 2);
        uint32_t b1 = (uint32_t)(((blockN + row1) * H_DIM + k0 + cc1) * 2);
        async_copy_b128(&As[buf][row0 * LDT + cc0], baseA, a0);
        async_copy_b128(&Bs[buf][row0 * LDT + cc0], baseB, b0);
        async_copy_b128(&As[buf][row1 * LDT + cc1], baseA, a1);
        async_copy_b128(&Bs[buf][row1 * LDT + cc1], baseB, b1);
        if (k0 + 64 < H_DIM) {   // warm L2 two slabs ahead
            __builtin_prefetch(&A[(blockM + row0) * H_DIM + k0 + cc0 + 64], 0, 1);
            __builtin_prefetch(&B[(blockN + row0) * H_DIM + k0 + cc0 + 64], 0, 1);
        }
    };

    issue_tile(0, 0);

    for (int it = 0; it < KTILES; ++it) {
        const int cur = it & 1;
        if (it + 1 < KTILES) {
            issue_tile((it + 1) * 32, cur ^ 1);
            // my 4 outstanding async ops belong to tile it+1; async loads
            // complete in order, so cnt<=4 means tile `it` has landed.
            asm volatile("s_wait_asynccnt 0x4" ::: "memory");
        } else {
            asm volatile("s_wait_asynccnt 0x0" ::: "memory");
        }
        __syncthreads();   // all waves' copies for tile `it` visible

        // B operands: lane<16 -> N=lr, K{0..7,16..23} ; lane>=16 -> N=lr, K{8..15,24..31}
        Frag bfrag[2];
#pragma unroll
        for (int nt = 0; nt < 2; ++nt) {
            const uint4* p = (const uint4*)&Bs[cur][(wn * 32 + nt * 16 + lr) * LDT + half * 8];
            bfrag[nt].q[0] = p[0];
            bfrag[nt].q[1] = p[2];
        }

#pragma unroll
        for (int mt = 0; mt < 4; ++mt) {
            // A operand: lane<16 -> M=lr, K{0..7,16..23}; lane>=16 -> K{8..15,24..31}
            Frag afrag;
            const uint16_t* ra = &As[cur][(wm * 64 + mt * 16 + lr) * LDT];
            afrag.q[0] = *(const uint4*)(ra + half * 8);
            afrag.q[1] = *(const uint4*)(ra + 16 + half * 8);
#pragma unroll
            for (int nt = 0; nt < 2; ++nt) {
                acc[mt][nt] = __builtin_amdgcn_wmma_f32_16x16x32_bf16(
                    false, afrag.v, false, bfrag[nt].v,
                    (short)0, acc[mt][nt], false, false);
            }
        }
        __syncthreads();   // buffer `cur` free for re-fill next iteration
    }

    // store: VGPR r -> M = base + r (+8 for lane half 1), N = base + lr
#pragma unroll
    for (int mt = 0; mt < 4; ++mt) {
#pragma unroll
        for (int nt = 0; nt < 2; ++nt) {
            int m0 = blockM + wm * 64 + mt * 16 + half * 8;
            int n  = blockN + wn * 32 + nt * 16 + lr;
#pragma unroll
            for (int r = 0; r < 8; ++r)
                C[(size_t)(m0 + r) * V_DIM + n] = acc[mt][nt][r];
        }
    }
}

// ---------------------------------------------------------------------------
// per-row JSD: online logsumexp for both distributions, then JSD sum
// one block (256 threads) per row
// ---------------------------------------------------------------------------
__device__ __forceinline__ float block_lse(const float* __restrict__ row,
                                           int tid, float* sm, float* ss) {
    float m = -INFINITY, s = 0.f;
    for (int j = tid; j < V_DIM; j += 256) {
        float x = row[j];
        float M = fmaxf(m, x);
        s = s * __expf(m - M) + __expf(x - M);
        m = M;
    }
    sm[tid] = m; ss[tid] = s;
    __syncthreads();
    for (int off = 128; off > 0; off >>= 1) {
        if (tid < off) {
            float m2 = sm[tid + off], s2 = ss[tid + off];
            float M  = fmaxf(sm[tid], m2);
            ss[tid]  = ss[tid] * __expf(sm[tid] - M) + s2 * __expf(m2 - M);
            sm[tid]  = M;
        }
        __syncthreads();
    }
    float r = sm[0] + __logf(ss[0]);
    __syncthreads();
    return r;
}

__global__ __launch_bounds__(256) void jsd_row(const float* __restrict__ sL,
                                               const float* __restrict__ tL,
                                               float* __restrict__ per_row) {
    __shared__ float sm[256], ss[256];
    const int b = blockIdx.x, tid = threadIdx.x;
    const size_t base = (size_t)b * V_DIM;

    float logZs = block_lse(sL + base, tid, sm, ss);
    float logZt = block_lse(tL + base, tid, sm, ss);

    float acc = 0.f;
    for (int j = tid; j < V_DIM; j += 256) {
        float slp = sL[base + j] - logZs;   // log Q
        float tlp = tL[base + j] - logZt;   // log P
        float Q = __expf(slp);
        float P = __expf(tlp);
        float Mv   = BETA_C * P + (1.f - BETA_C) * Q;
        float logM = __logf(Mv);
        acc += BETA_C * P * (tlp - logM) + (1.f - BETA_C) * Q * (slp - logM);
    }
    sm[tid] = acc;
    __syncthreads();
    for (int off = 128; off > 0; off >>= 1) {
        if (tid < off) sm[tid] += sm[tid + off];
        __syncthreads();
    }
    if (tid == 0) per_row[b] = sm[0];
}

// ---------------------------------------------------------------------------
// deterministic finalize: masked mean over rows
// ---------------------------------------------------------------------------
__global__ __launch_bounds__(256) void finalize(const float* __restrict__ per_row,
                                                const int* __restrict__ labels,
                                                float* __restrict__ out) {
    __shared__ float s[256];
    __shared__ int   c[256];
    const int tid = threadIdx.x;
    float acc = 0.f; int cnt = 0;
    for (int b = tid; b < BT_DIM; b += 256) {
        if (labels[b] != IGNORE_INDEX) { acc += per_row[b]; ++cnt; }
    }
    s[tid] = acc; c[tid] = cnt;
    __syncthreads();
    for (int off = 128; off > 0; off >>= 1) {
        if (tid < off) { s[tid] += s[tid + off]; c[tid] += c[tid + off]; }
        __syncthreads();
    }
    if (tid == 0) {
        int n = c[0] > 1 ? c[0] : 1;
        out[0] = s[0] / (float)n;
    }
}

// ---------------------------------------------------------------------------
extern "C" void kernel_launch(void* const* d_in, const int* in_sizes, int n_in,
                              void* d_out, int out_size, void* d_ws, size_t ws_size,
                              hipStream_t stream) {
    const float* sX = (const float*)d_in[0];
    const float* sW = (const float*)d_in[1];
    const float* tX = (const float*)d_in[2];
    const float* tW = (const float*)d_in[3];
    const int* labels = (const int*)d_in[4];
    float* out = (float*)d_out;

    char* w = (char*)d_ws;
    size_t off = 0;
    auto alloc = [&](size_t bytes) -> void* {
        void* p = w + off;
        off = (off + bytes + 255) & ~(size_t)255;
        return p;
    };
    uint16_t* sXb = (uint16_t*)alloc((size_t)BT_DIM * H_DIM * 2);
    uint16_t* tXb = (uint16_t*)alloc((size_t)BT_DIM * H_DIM * 2);
    uint16_t* sWb = (uint16_t*)alloc((size_t)V_DIM * H_DIM * 2);
    uint16_t* tWb = (uint16_t*)alloc((size_t)V_DIM * H_DIM * 2);
    float* sL = (float*)alloc((size_t)BT_DIM * V_DIM * 4);
    float* tL = (float*)alloc((size_t)BT_DIM * V_DIM * 4);
    float* per_row = (float*)alloc((size_t)BT_DIM * 4);

    const int nX4 = BT_DIM * H_DIM / 4;
    const int nW4 = V_DIM * H_DIM / 4;
    cvt_f32_bf16<<<(nX4 + 255) / 256, 256, 0, stream>>>(sX, sXb, nX4);
    cvt_f32_bf16<<<(nX4 + 255) / 256, 256, 0, stream>>>(tX, tXb, nX4);
    cvt_f32_bf16<<<(nW4 + 255) / 256, 256, 0, stream>>>(sW, sWb, nW4);
    cvt_f32_bf16<<<(nW4 + 255) / 256, 256, 0, stream>>>(tW, tWb, nW4);

    gemm_bf16<<<dim3(V_DIM / 128, BT_DIM / 128, 2), 256, 0, stream>>>(
        sXb, sWb, sL, tXb, tWb, tL);

    jsd_row<<<BT_DIM, 256, 0, stream>>>(sL, tL, per_row);
    finalize<<<1, 256, 0, stream>>>(per_row, labels, out);
}
